// CrossAttention_26731876450618
// MI455X (gfx1250) — compile-verified, matrix operand order checked by reading, not run on hardware
//
#include <hip/hip_runtime.h>
#include <hip/hip_bf16.h>
#include <math.h>

// ---------------- types ----------------
typedef __bf16 bf16;
typedef __bf16 bf16x16 __attribute__((ext_vector_type(16)));
typedef __bf16 bf16x8  __attribute__((ext_vector_type(8)));
typedef __bf16 bf16x4  __attribute__((ext_vector_type(4)));
typedef float  f32x8   __attribute__((ext_vector_type(8)));

#define BB 4
#define TT 1024
#define HH 16
#define HD 64
#define EE 1024
#define DD 1025
#define DP 1056            // padded K (33 tiles of 32), zero-filled tail
#define MROWS 4096         // B*T
#define EPSV 1e-8f

// CDNA5 async global->LDS copy (16B per lane), tracked by ASYNCcnt.
// dst: per-lane LDS byte address (low 32 bits of the flat shared pointer),
// src: per-lane 64-bit global address.
#define ASYNC_CP_B128(ldsoff, gptr)                                         \
    asm volatile("global_load_async_to_lds_b128 %0, %1, off"                \
                 :: "v"(ldsoff), "v"(gptr) : "memory")
#define WAIT_ASYNCCNT0() asm volatile("s_wait_asynccnt 0" ::: "memory")

static __device__ __forceinline__ unsigned lds_addr(const void* p)
{
    return (unsigned)(unsigned long long)p;
}

// =====================================================================
// f32[rows][1025] -> bf16[rows][1056], zero-padded columns 1025..1055.
// =====================================================================
__global__ __launch_bounds__(256) void cvt_bf16_pad(const float* __restrict__ src,
                                                    bf16* __restrict__ dst)
{
    const int r = blockIdx.x, tid = threadIdx.x;
    const float* sp = src + (size_t)r * DD;
    bf16* dp = dst + (size_t)r * DP;
    const int c0 = tid * 4;                 // covers 0..1023
    bf16x4 v;
    for (int j = 0; j < 4; ++j) v[j] = (bf16)sp[c0 + j];
    *(bf16x4*)(dp + c0) = v;
    if (tid < 32) {                         // covers 1024..1055 (pad)
        int c = 1024 + tid;
        dp[c] = (bf16)((c < DD) ? sp[c] : 0.f);
    }
}

// =====================================================================
// GEMM: C[M=4096, N=1024] = A[M, K] @ W[1..1024, K]^T + bias[1..1024]
// A, W pre-converted bf16 (row stride DP). 128x128 tile, one WMMA
// k-step (32) per K-tile. Staging via async global->LDS b128 copies.
// =====================================================================
__global__ __launch_bounds__(256) void gemm_space(const bf16* __restrict__ A,
                                                  const bf16* __restrict__ W,
                                                  const float* __restrict__ bias,
                                                  float* __restrict__ C)
{
    // row stride 40 halves = 80 bytes (16B multiple -> b128-friendly)
    __shared__ bf16 sA[128 * 40];
    __shared__ bf16 sB[128 * 40];

    const int tid  = threadIdx.x;
    const int lane = tid & 31;
    const int wave = tid >> 5;
    const int m0 = blockIdx.y * 128;
    const int n0 = blockIdx.x * 128;
    const int wm = (wave & 3) * 32;     // wave row base within tile
    const int wn = (wave >> 2) * 64;    // wave col base within tile
    const int lm = lane & 15;
    const int hi = lane >> 4;

    f32x8 acc[2][4] = {};

    const int ar = tid & 127;            // loader row in tile
    const int ac = (tid >> 7) * 16;      // loader col base (0 or 16)
    const bf16* gA = A + (size_t)(m0 + ar) * DP + ac;
    const bf16* gW = W + (size_t)(n0 + ar + 1) * DP + ac;
    const unsigned dA = lds_addr(&sA[ar * 40 + ac]);
    const unsigned dB = lds_addr(&sB[ar * 40 + ac]);

    for (int k0 = 0; k0 < DP; k0 += 32) {
        if (k0 + 32 < DP) {                       // pull next tile toward L0/L2
            __builtin_prefetch(gA + k0 + 32, 0, 3);
            __builtin_prefetch(gW + k0 + 32, 0, 3);
        }
        // async stage this K-tile (16B x 4 per thread), ASYNCcnt-tracked
        ASYNC_CP_B128(dA,      gA + k0);
        ASYNC_CP_B128(dA + 16, gA + k0 + 8);
        ASYNC_CP_B128(dB,      gW + k0);
        ASYNC_CP_B128(dB + 16, gW + k0 + 8);
        WAIT_ASYNCCNT0();
        __syncthreads();

        bf16x16 afr[2];
        for (int mi = 0; mi < 2; ++mi) {
            const bf16* p = &sA[(wm + mi * 16 + lm) * 40 + hi * 8];
            union { bf16x16 v; bf16x8 q[2]; } u;
            u.q[0] = *(const bf16x8*)p;          // k = hi*8 .. +7
            u.q[1] = *(const bf16x8*)(p + 16);   // k = 16+hi*8 .. +7
            afr[mi] = u.v;
        }
        for (int ni = 0; ni < 4; ++ni) {
            const bf16* p = &sB[(wn + ni * 16 + lm) * 40 + hi * 8];
            union { bf16x16 v; bf16x8 q[2]; } u;
            u.q[0] = *(const bf16x8*)p;
            u.q[1] = *(const bf16x8*)(p + 16);
            for (int mi = 0; mi < 2; ++mi)
                acc[mi][ni] = __builtin_amdgcn_wmma_f32_16x16x32_bf16(
                    false, afr[mi], false, u.v, (short)0, acc[mi][ni], false, false);
        }
        __syncthreads();
    }

    for (int ni = 0; ni < 4; ++ni) {
        int n = n0 + wn + ni * 16 + lm;
        float bv = bias[n + 1];
        for (int mi = 0; mi < 2; ++mi)
            for (int r = 0; r < 8; ++r) {
                int m = m0 + wm + mi * 16 + r + 8 * hi;
                C[(size_t)m * EE + n] = acc[mi][ni][r] + bv;
            }
    }
}

// =====================================================================
// lorentz_linear epilogue + shape_heads.  One block per row m.
// =====================================================================
__global__ __launch_bounds__(256) void qkv_finish(const float* __restrict__ H,
                                                  const float* __restrict__ Ain,
                                                  const float* __restrict__ W0,
                                                  const float* __restrict__ bias,
                                                  const float* __restrict__ log_scale,
                                                  bf16* __restrict__ Sout,
                                                  float* __restrict__ Tout)
{
    __shared__ float red[256];
    __shared__ float red2[256];
    __shared__ float hpart[256];
    __shared__ float hsum[16];
    __shared__ float sc_s;

    const int m = blockIdx.x;
    const int b = m >> 10, t = m & 1023;
    const int tid = threadIdx.x;

    float hv[4];
    float ssq = 0.f;
    for (int j = 0; j < 4; ++j) {
        hv[j] = H[(size_t)m * EE + tid * 4 + j];
        ssq += hv[j] * hv[j];
    }
    float td = 0.f;
    for (int p = 0; p < 5; ++p) {
        int i = tid + p * 256;
        if (i < DD) td += Ain[(size_t)m * DD + i] * W0[i];
    }
    red[tid] = ssq; red2[tid] = td; hpart[tid] = ssq;
    __syncthreads();
    for (int s = 128; s > 0; s >>= 1) {
        if (tid < s) { red[tid] += red[tid + s]; red2[tid] += red2[tid + s]; }
        __syncthreads();
    }
    if (tid == 0) {
        float s2 = fmaxf(red[0], EPSV);
        float ht = red2[0] + bias[0];
        float tm = 1.f / (1.f + __expf(-ht)) * __expf(log_scale[0]) + 1.1f;
        sc_s = sqrtf((tm * tm - 1.f) / s2);
    }
    if (tid < 16) {
        float hs = 0.f;
        for (int j = 0; j < 16; ++j) hs += hpart[tid * 16 + j];
        hsum[tid] = hs;
    }
    __syncthreads();
    float sc = sc_s;
    const int head = tid >> 4;
    if ((tid & 15) == 0)
        Tout[((size_t)(b * HH + head)) * TT + t] = sqrtf(1.f + sc * sc * hsum[head]);
    size_t base = (((size_t)(b * HH + head)) * TT + t) * (size_t)HD + (tid & 15) * 4;
    for (int j = 0; j < 4; ++j) Sout[base + j] = (bf16)(hv[j] * sc);
}

// =====================================================================
// Flash-style Lorentz attention. Block = 128 threads (4 waves),
// one block handles (b,h) x 64 query rows; streams 128-key tiles.
// K tile staged via async global->LDS; V staged transposed.
// =====================================================================
__global__ __launch_bounds__(128) void lorentz_attn(const bf16* __restrict__ Qs,
                                                    const float* __restrict__ Qt,
                                                    const bf16* __restrict__ Ks,
                                                    const float* __restrict__ Kt,
                                                    const bf16* __restrict__ Vs,
                                                    const float* __restrict__ Vt,
                                                    const float* __restrict__ abias_p,
                                                    float* __restrict__ Xo)
{
    __shared__ bf16 sK[128 * 72];        // [s][d], stride 144B
    __shared__ bf16 sV[64 * 136];        // transposed [d][s], stride 272B
    __shared__ bf16 sP[4 * 16 * 136];    // per-wave P tile [m][s]
    __shared__ float sKt[128];
    __shared__ float sVt[128];

    const int tid  = threadIdx.x;
    const int lane = tid & 31;
    const int wave = tid >> 5;
    const int lm = lane & 15;
    const int hi = lane >> 4;
    const int bh = blockIdx.y;                 // b*16 + h
    const int b  = bh >> 4, hh = bh & 15;
    const int t0 = blockIdx.x * 64;
    const int trow = t0 + wave * 16;
    const size_t hbase = (size_t)bh * TT;

    // Q fragments (A-matrix layout), two 32-wide k-steps over HD=64
    bf16x16 qfr[2];
    {
        const bf16* qp = Qs + (hbase + trow + lm) * HD;
        for (int ks = 0; ks < 2; ++ks) {
            union { bf16x16 v; bf16x8 q[2]; } u;
            u.q[0] = *(const bf16x8*)(qp + ks * 32 + hi * 8);
            u.q[1] = *(const bf16x8*)(qp + ks * 32 + 16 + hi * 8);
            qfr[ks] = u.v;
        }
    }
    float qtv[8];
    for (int r = 0; r < 8; ++r) qtv[r] = Qt[hbase + trow + r + 8 * hi];

    f32x8 oacc[4] = {};
    float mrun[8], lrun[8], avet[8];
    for (int r = 0; r < 8; ++r) { mrun[r] = -3.0e38f; lrun[r] = 0.f; avet[r] = 0.f; }
    const float abias = abias_p[0];

    const unsigned dKrow = lds_addr(&sK[tid * 72]);

    for (int s0 = 0; s0 < TT; s0 += 128) {
        if (s0 + 128 < TT) {                       // prefetch next key/value tile
            __builtin_prefetch(Ks + (hbase + s0 + 128 + tid) * HD, 0, 3);
            __builtin_prefetch(Vs + (hbase + s0 + 128 + tid) * HD, 0, 3);
        }
        {
            const bf16* kp = Ks + (hbase + s0 + tid) * HD;
            const bf16* vp = Vs + (hbase + s0 + tid) * HD;
            for (int c = 0; c < 8; ++c)            // async K row copy (8 x 16B)
                ASYNC_CP_B128(dKrow + c * 16, kp + c * 8);
            for (int c = 0; c < HD; ++c)           // V transpose (scatter b16)
                sV[c * 136 + tid] = vp[c];
            sKt[tid] = Kt[hbase + s0 + tid];
            sVt[tid] = Vt[hbase + s0 + tid];
        }
        WAIT_ASYNCCNT0();
        __syncthreads();

        // S = Q . K^T (space part) via WMMA
        f32x8 sacc[8];
        for (int nt = 0; nt < 8; ++nt) {
            f32x8 c = {};
            for (int ks = 0; ks < 2; ++ks) {
                const bf16* p = &sK[(nt * 16 + lm) * 72 + ks * 32 + hi * 8];
                union { bf16x16 v; bf16x8 q[2]; } u;
                u.q[0] = *(const bf16x8*)p;
                u.q[1] = *(const bf16x8*)(p + 16);
                c = __builtin_amdgcn_wmma_f32_16x16x32_bf16(
                        false, qfr[ks], false, u.v, (short)0, c, false, false);
            }
            sacc[nt] = c;
        }

        // logits with Lorentz time term, row max
        float rowmax[8];
        for (int r = 0; r < 8; ++r) rowmax[r] = -3.0e38f;
        for (int nt = 0; nt < 8; ++nt) {
            float ktv = sKt[nt * 16 + lm];
            for (int r = 0; r < 8; ++r) {
                float cin = sacc[nt][r] - qtv[r] * ktv;
                float lg = (2.f + 2.f * cin) * 0.125f + abias;
                sacc[nt][r] = lg;
                rowmax[r] = fmaxf(rowmax[r], lg);
            }
        }
        for (int r = 0; r < 8; ++r) {
            float v = rowmax[r];
            for (int off = 1; off < 16; off <<= 1)
                v = fmaxf(v, __shfl_xor(v, off, 32));
            rowmax[r] = v;
        }

        // online softmax
        float alpha[8], rsum[8], avadd[8];
        for (int r = 0; r < 8; ++r) {
            float mnew = fmaxf(mrun[r], rowmax[r]);
            alpha[r] = __expf(mrun[r] - mnew);
            mrun[r] = mnew;
            rsum[r] = 0.f; avadd[r] = 0.f;
        }
        for (int nt = 0; nt < 8; ++nt) {
            float vtv = sVt[nt * 16 + lm];
            for (int r = 0; r < 8; ++r) {
                float p = __expf(sacc[nt][r] - mrun[r]);
                sacc[nt][r] = p;
                rsum[r] += p;
                avadd[r] += p * vtv;
            }
        }
        for (int r = 0; r < 8; ++r) {
            float v = rsum[r];
            for (int off = 1; off < 16; off <<= 1) v += __shfl_xor(v, off, 32);
            lrun[r] = lrun[r] * alpha[r] + v;
            avet[r] = avet[r] * alpha[r] + avadd[r];   // lane-partial, reduced at end
        }
        for (int dt = 0; dt < 4; ++dt)
            for (int r = 0; r < 8; ++r)
                oacc[dt][r] *= alpha[r];

        // stage P (bf16) in this wave's LDS region (A-layout transpose)
        bf16* pw = &sP[wave * 16 * 136];
        for (int nt = 0; nt < 8; ++nt)
            for (int r = 0; r < 8; ++r)
                pw[(r + 8 * hi) * 136 + nt * 16 + lm] = (bf16)sacc[nt][r];
        __syncthreads();

        // O += P . V via WMMA (4 k-steps over 128 keys)
        for (int ks = 0; ks < 4; ++ks) {
            union { bf16x16 v; bf16x8 q[2]; } ua;
            const bf16* pa = &pw[lm * 136 + ks * 32 + hi * 8];
            ua.q[0] = *(const bf16x8*)pa;
            ua.q[1] = *(const bf16x8*)(pa + 16);
            for (int dt = 0; dt < 4; ++dt) {
                const bf16* pb = &sV[(dt * 16 + lm) * 136 + ks * 32 + hi * 8];
                union { bf16x16 v; bf16x8 q[2]; } ub;
                ub.q[0] = *(const bf16x8*)pb;
                ub.q[1] = *(const bf16x8*)(pb + 16);
                oacc[dt] = __builtin_amdgcn_wmma_f32_16x16x32_bf16(
                        false, ua.v, false, ub.v, (short)0, oacc[dt], false, false);
            }
        }
        __syncthreads();
    }

    // finalize: ave = O/l, Lorentz-normalize per row, scatter into xo space cols
    for (int r = 0; r < 8; ++r) {
        float inv = 1.f / lrun[r];
        float sp = 0.f;
        for (int dt = 0; dt < 4; ++dt) {
            float v = oacc[dt][r] * inv;
            oacc[dt][r] = v;
            sp += v * v;
        }
        float at = avet[r] * inv;
        for (int off = 1; off < 16; off <<= 1) {
            sp += __shfl_xor(sp, off, 32);
            at += __shfl_xor(at, off, 32);
        }
        float lin = fabsf(sp - at * at);
        float rden = 1.f / sqrtf(fmaxf(lin, EPSV));
        int t = trow + r + 8 * hi;
        size_t rowb = ((size_t)(b * TT + t)) * DD + 1 + hh * HD;
        for (int dt = 0; dt < 4; ++dt)
            Xo[rowb + dt * 16 + lm] = oacc[dt][r] * rden;
    }
}

// ===================== xo[:,0] = sqrt(1 + ||space||^2) =====================
__global__ __launch_bounds__(256) void row_time(float* __restrict__ X)
{
    __shared__ float red[256];
    const int m = blockIdx.x, tid = threadIdx.x;
    float s = 0.f;
    for (int j = 0; j < 4; ++j) {
        float v = X[(size_t)m * DD + 1 + tid * 4 + j];
        s += v * v;
    }
    red[tid] = s; __syncthreads();
    for (int k = 128; k > 0; k >>= 1) {
        if (tid < k) red[tid] += red[tid + k];
        __syncthreads();
    }
    if (tid == 0) X[(size_t)m * DD] = sqrtf(1.f + red[0]);
}

// ===================== final lorentz_linear epilogue =====================
__global__ __launch_bounds__(256) void out_finish(const float* __restrict__ H,
                                                  const float* __restrict__ Ain,
                                                  const float* __restrict__ W0,
                                                  const float* __restrict__ bias,
                                                  const float* __restrict__ log_scale,
                                                  float* __restrict__ Out)
{
    __shared__ float red[256];
    __shared__ float red2[256];
    __shared__ float sc_s, t_s;
    const int m = blockIdx.x, tid = threadIdx.x;
    float hv[4]; float ssq = 0.f;
    for (int j = 0; j < 4; ++j) {
        hv[j] = H[(size_t)m * EE + tid * 4 + j];
        ssq += hv[j] * hv[j];
    }
    float td = 0.f;
    for (int p = 0; p < 5; ++p) {
        int i = tid + p * 256;
        if (i < DD) td += Ain[(size_t)m * DD + i] * W0[i];
    }
    red[tid] = ssq; red2[tid] = td; __syncthreads();
    for (int s = 128; s > 0; s >>= 1) {
        if (tid < s) { red[tid] += red[tid + s]; red2[tid] += red2[tid + s]; }
        __syncthreads();
    }
    if (tid == 0) {
        float s2 = fmaxf(red[0], EPSV);
        float ht = red2[0] + bias[0];
        float tm = 1.f / (1.f + __expf(-ht)) * __expf(log_scale[0]) + 1.1f;
        t_s = tm;
        sc_s = sqrtf((tm * tm - 1.f) / s2);
    }
    __syncthreads();
    if (tid == 0) Out[(size_t)m * DD] = t_s;
    float sc = sc_s;
    for (int j = 0; j < 4; ++j)
        Out[(size_t)m * DD + 1 + tid * 4 + j] = hv[j] * sc;
}

// =====================================================================
extern "C" void kernel_launch(void* const* d_in, const int* in_sizes, int n_in,
                              void* d_out, int out_size, void* d_ws, size_t ws_size,
                              hipStream_t stream)
{
    const float* x   = (const float*)d_in[0];
    const float* y   = (const float*)d_in[1];
    const float* Wq  = (const float*)d_in[2];
    const float* bq  = (const float*)d_in[3];
    const float* sq  = (const float*)d_in[4];
    const float* Wk  = (const float*)d_in[5];
    const float* bk  = (const float*)d_in[6];
    const float* sk  = (const float*)d_in[7];
    const float* Wv  = (const float*)d_in[8];
    const float* bv  = (const float*)d_in[9];
    const float* sv  = (const float*)d_in[10];
    const float* Wo  = (const float*)d_in[11];
    const float* bo  = (const float*)d_in[12];
    const float* so  = (const float*)d_in[13];
    const float* ab  = (const float*)d_in[14];
    float* out = (float*)d_out;

    char* w = (char*)d_ws;
    float* h_tmp = (float*)w; w += (size_t)MROWS * EE * 4;         // 16.78 MB
    float* xo    = (float*)w; w += (size_t)MROWS * DD * 4;         // 16.79 MB
    bf16*  qs    = (bf16*)w;  w += (size_t)BB * HH * TT * HD * 2;  // 8.39 MB
    bf16*  ksb   = (bf16*)w;  w += (size_t)BB * HH * TT * HD * 2;
    bf16*  vsb   = (bf16*)w;  w += (size_t)BB * HH * TT * HD * 2;
    float* qt    = (float*)w; w += (size_t)BB * HH * TT * 4;
    float* kt    = (float*)w; w += (size_t)BB * HH * TT * 4;
    float* vt    = (float*)w; w += (size_t)BB * HH * TT * 4;
    bf16*  xbf   = (bf16*)w;  w += (size_t)MROWS * DP * 2;         // 8.65 MB
    bf16*  ybf   = (bf16*)w;  w += (size_t)MROWS * DP * 2;
    bf16*  xobf  = (bf16*)w;  w += (size_t)MROWS * DP * 2;
    bf16*  wqbf  = (bf16*)w;  w += (size_t)DD * DP * 2;            // 2.16 MB
    bf16*  wkbf  = (bf16*)w;  w += (size_t)DD * DP * 2;
    bf16*  wvbf  = (bf16*)w;  w += (size_t)DD * DP * 2;
    bf16*  wobf  = (bf16*)w;  w += (size_t)DD * DP * 2;

    // one-time bf16 conversion / padding of GEMM operands
    cvt_bf16_pad<<<MROWS, 256, 0, stream>>>(x,  xbf);
    cvt_bf16_pad<<<MROWS, 256, 0, stream>>>(y,  ybf);
    cvt_bf16_pad<<<DD,    256, 0, stream>>>(Wq, wqbf);
    cvt_bf16_pad<<<DD,    256, 0, stream>>>(Wk, wkbf);
    cvt_bf16_pad<<<DD,    256, 0, stream>>>(Wv, wvbf);
    cvt_bf16_pad<<<DD,    256, 0, stream>>>(Wo, wobf);

    dim3 gGemm(EE / 128, MROWS / 128);   // (8, 32)

    // Q projection
    gemm_space<<<gGemm, 256, 0, stream>>>(xbf, wqbf, bq, h_tmp);
    qkv_finish<<<MROWS, 256, 0, stream>>>(h_tmp, x, Wq, bq, sq, qs, qt);
    // K projection
    gemm_space<<<gGemm, 256, 0, stream>>>(ybf, wkbf, bk, h_tmp);
    qkv_finish<<<MROWS, 256, 0, stream>>>(h_tmp, y, Wk, bk, sk, ksb, kt);
    // V projection
    gemm_space<<<gGemm, 256, 0, stream>>>(ybf, wvbf, bv, h_tmp);
    qkv_finish<<<MROWS, 256, 0, stream>>>(h_tmp, y, Wv, bv, sv, vsb, vt);
    // Attention -> xo space columns
    lorentz_attn<<<dim3(TT / 64, BB * HH), 128, 0, stream>>>(qs, qt, ksb, kt, vsb, vt, ab, xo);
    // xo time column, then bf16 staging for the final GEMM
    row_time<<<MROWS, 256, 0, stream>>>(xo);
    cvt_bf16_pad<<<MROWS, 256, 0, stream>>>(xo, xobf);
    // Output projection
    gemm_space<<<gGemm, 256, 0, stream>>>(xobf, wobf, bo, h_tmp);
    out_finish<<<MROWS, 256, 0, stream>>>(h_tmp, xo, Wo, bo, so, out);
}
